// DeepFMRankModel_21844203668198
// MI455X (gfx1250) — matrix operand order repaired and checked by reading, not compiled, not run on hardware
//
#include <hip/hip_runtime.h>
#include <hip/hip_bf16.h>

typedef __attribute__((ext_vector_type(16))) _Float16 v16h;
typedef __attribute__((ext_vector_type(8)))  _Float16 v8h;
typedef __attribute__((ext_vector_type(8)))  float    v8f;

#define N_TABLES 26
#define VOCAB    100000
#define EMBED    16
#define F_FLOAT  13
#define HIDDEN   400
#define BATCH    16384
#define B2       (2*BATCH)
#define KV       416   /* 26*16 : GEMM1 K (V_f part folded into bias) */
#define KH       416   /* HIDDEN padded to 13*32 for GEMM2 */
#define MB       4     /* M-tiles per wave: reuse B fragment across 4 WMMAs */

__device__ __forceinline__ float sigf(float x) { return 1.0f / (1.0f + __expf(-x)); }

// ---- W2 (624x400 f32, row-major) -> W2h (400x416 f16, n-major over first 416 k) ----
__global__ __launch_bounds__(256) void k_w2(const float* __restrict__ W2, _Float16* __restrict__ W2h) {
    int t = blockIdx.x * 256 + threadIdx.x;
    if (t >= HIDDEN * KV) return;
    int n = t / KV, k = t % KV;
    W2h[t] = (_Float16)W2[(size_t)k * HIDDEN + n];
}

// ---- W3 (400x400) -> W3h (400x416 f16, zero pad k>=400) ----
__global__ __launch_bounds__(256) void k_w3(const float* __restrict__ W3, _Float16* __restrict__ W3h) {
    int t = blockIdx.x * 256 + threadIdx.x;
    if (t >= HIDDEN * KH) return;
    int n = t / KH, k = t % KH;
    W3h[t] = (k < HIDDEN) ? (_Float16)W3[(size_t)k * HIDDEN + n] : (_Float16)0.0f;
}

// ---- b2eff[n] = b2[n] + sum_m Vf_flat[m] * W2[416+m][n]  (V_f rows are batch-constant) ----
__global__ __launch_bounds__(256) void k_b2eff(const float* __restrict__ W2, const float* __restrict__ b2,
                                               const float* __restrict__ Vf, float* __restrict__ b2eff) {
    int n = blockIdx.x * 256 + threadIdx.x;
    if (n >= HIDDEN) return;
    float acc = b2[n];
    for (int m = 0; m < F_FLOAT * EMBED; ++m)
        acc += Vf[m] * W2[(size_t)(KV + m) * HIDDEN + n];
    b2eff[n] = acc;
}

// ---- Embedding gather + FM (linear + interaction) + write A-matrix for GEMM1 ----
__global__ __launch_bounds__(256) void k_embed(const int* __restrict__ xi_p, const float* __restrict__ xf_p,
                                               const int* __restrict__ xi_n, const float* __restrict__ xf_n,
                                               const float* __restrict__ emb, const float* __restrict__ W1,
                                               const float* __restrict__ b1, const float* __restrict__ Vf,
                                               _Float16* __restrict__ Vh, float* __restrict__ yfm) {
    int i = blockIdx.x * 256 + threadIdx.x;
    if (i >= B2) return;
    int idx = i & (BATCH - 1);
    const int*   xi = (i < BATCH) ? xi_p : xi_n;
    const float* xf = (i < BATCH) ? xf_p : xf_n;

    float s[EMBED], ss[EMBED];
#pragma unroll
    for (int k = 0; k < EMBED; ++k) { s[k] = 0.f; ss[k] = 0.f; }

    float lin = b1[0];
    float xv[F_FLOAT];
#pragma unroll
    for (int j = 0; j < F_FLOAT; ++j) {
        xv[j] = xf[(size_t)idx * F_FLOAT + j];
        lin += xv[j] * W1[j];
    }

    _Float16* vrow = Vh + (size_t)i * KV;
    for (int t = 0; t < N_TABLES; ++t) {
        int x = xi[(size_t)idx * N_TABLES + t];
        const float* row = emb + ((size_t)t * VOCAB + (size_t)x) * (EMBED + 1);
        lin += row[0];
#pragma unroll
        for (int k = 0; k < EMBED; ++k) {
            float v = row[1 + k];
            s[k] += v; ss[k] += v * v;
            vrow[t * EMBED + k] = (_Float16)v;
        }
    }
    for (int j = 0; j < F_FLOAT; ++j) {
        float w = xv[j];
#pragma unroll
        for (int k = 0; k < EMBED; ++k) {
            float v = Vf[j * EMBED + k] * w;
            s[k] += v; ss[k] += v * v;
        }
    }
    float inter = 0.f;
#pragma unroll
    for (int k = 0; k < EMBED; ++k) inter += s[k] * s[k] - ss[k];
    yfm[i] = lin + 0.5f * inter;
}

// ---- zero-pad H1 columns 400..415 so GEMM2 K=416 reduction is exact ----
__global__ __launch_bounds__(256) void k_padh1(_Float16* __restrict__ H1h) {
    int i = blockIdx.x * 256 + threadIdx.x;
    if (i >= B2) return;
    v8h z;
#pragma unroll
    for (int j = 0; j < 8; ++j) z[j] = (_Float16)0.0f;
    *(v8h*)(H1h + (size_t)i * KH + HIDDEN)     = z;
    *(v8h*)(H1h + (size_t)i * KH + HIDDEN + 8) = z;
}

// ---- WMMA GEMM: C[m][n] = swish(sum_k A[m][k]*B[n][k] + bias[n]) ----
// Each wave: 64x16 output (4 M-tiles sharing one B fragment per k-step,
// 4 independent accumulator chains to hide WMMA->WMMA latency).
__global__ __launch_bounds__(256) void k_gemm_swish(const _Float16* __restrict__ A, int lda,
                                                    const _Float16* __restrict__ B, int ldb,
                                                    const float* __restrict__ bias,
                                                    _Float16* __restrict__ C, int ldc,
                                                    int ksteps, int ntiles, int mblocks) {
    int wave = (blockIdx.x * 256 + threadIdx.x) >> 5;
    int lane = threadIdx.x & 31;
    int mb = wave / ntiles;
    int nt = wave - mb * ntiles;
    if (mb >= mblocks) return;   // wave-uniform exit; live waves keep EXEC all-ones
    int m0 = mb * (16 * MB), n0 = nt * 16;
    int half = lane >> 4, lr = lane & 15;

    // A: lane holds row m+lr, K chunk [k0+16*half .. +15] contiguous
    const _Float16* arow0 = A + (size_t)(m0 + lr) * lda + half * 16;
    // B (stored N-major): lane holds column n0+lr, K chunk [k0+16*half .. +15] contiguous
    const _Float16* brow = B + (size_t)(n0 + lr) * ldb + half * 16;

    v8f c[MB];
#pragma unroll
    for (int t = 0; t < MB; ++t)
#pragma unroll
        for (int r = 0; r < 8; ++r) c[t][r] = 0.f;

    for (int ks = 0; ks < ksteps; ++ks) {
        v16h b = *(const v16h*)(brow + ks * 32);
#pragma unroll
        for (int t = 0; t < MB; ++t) {
            const _Float16* ap = arow0 + (size_t)(t * 16) * lda + ks * 32;
            v8h alo = *(const v8h*)(ap);
            v8h ahi = *(const v8h*)(ap + 8);
            v16h a;
#pragma unroll
            for (int j = 0; j < 8; ++j) { a[j] = alo[j]; a[8 + j] = ahi[j]; }
            c[t] = __builtin_amdgcn_wmma_f32_16x16x32_f16(false, a, false, b, (short)0, c[t], false, false);
        }
    }

#pragma unroll
    for (int t = 0; t < MB; ++t) {
#pragma unroll
        for (int r = 0; r < 8; ++r) {
            int m = m0 + t * 16 + r + half * 8;
            int n = n0 + lr;
            float h = c[t][r] + bias[n];
            C[(size_t)m * ldc + n] = (_Float16)(h * sigf(h));
        }
    }
}

// ---- final: 400-dot with W4, swish, + y_fm, sigmoid(pos)-sigmoid(neg) ----
__global__ __launch_bounds__(256) void k_final(const _Float16* __restrict__ H2h, const float* __restrict__ yfm,
                                               const float* __restrict__ W4, const float* __restrict__ b4,
                                               float* __restrict__ out) {
    int p = blockIdx.x * 256 + threadIdx.x;
    if (p >= BATCH) return;
    float r[2];
#pragma unroll
    for (int sdx = 0; sdx < 2; ++sdx) {
        int i = p + sdx * BATCH;
        const _Float16* hrow = H2h + (size_t)i * HIDDEN;
        float acc = b4[0];
        for (int k = 0; k < HIDDEN; ++k) acc += (float)hrow[k] * W4[k];
        r[sdx] = yfm[i] + acc * sigf(acc);
    }
    out[p] = sigf(r[0]) - sigf(r[1]);
}

extern "C" void kernel_launch(void* const* d_in, const int* in_sizes, int n_in,
                              void* d_out, int out_size, void* d_ws, size_t ws_size,
                              hipStream_t stream) {
    const int*   xi_p = (const int*)d_in[0];
    const float* xf_p = (const float*)d_in[1];
    const int*   xi_n = (const int*)d_in[2];
    const float* xf_n = (const float*)d_in[3];
    const float* emb  = (const float*)d_in[4];
    const float* W1   = (const float*)d_in[5];
    const float* b1   = (const float*)d_in[6];
    const float* Vf   = (const float*)d_in[7];
    const float* W2   = (const float*)d_in[8];
    const float* b2   = (const float*)d_in[9];
    const float* W3   = (const float*)d_in[10];
    const float* b3   = (const float*)d_in[11];
    const float* W4   = (const float*)d_in[12];
    const float* b4   = (const float*)d_in[13];
    float* out = (float*)d_out;

    char* ws = (char*)d_ws;
    // offsets (all 256B-aligned; f16 row strides 832B keep 16B/32B vector alignment)
    _Float16* W2h   = (_Float16*)(ws + 0);         // 400*416*2 = 332800
    _Float16* W3h   = (_Float16*)(ws + 332800);    // 332800
    float*    b2eff = (float*)   (ws + 665600);    // 1600 -> pad 1792
    float*    yfm   = (float*)   (ws + 667392);    // 32768*4 = 131072
    _Float16* Vh    = (_Float16*)(ws + 798464);    // 32768*416*2 = 27262976 (reused as H2h)
    _Float16* H1h   = (_Float16*)(ws + 28061440);  // 32768*416*2 = 27262976 ; end = 55324416
    _Float16* H2h   = Vh;                          // V no longer needed after GEMM1

    k_w2   <<<(HIDDEN*KV + 255)/256, 256, 0, stream>>>(W2, W2h);
    k_w3   <<<(HIDDEN*KH + 255)/256, 256, 0, stream>>>(W3, W3h);
    k_b2eff<<<2, 256, 0, stream>>>(W2, b2, Vf, b2eff);
    k_embed<<<B2/256, 256, 0, stream>>>(xi_p, xf_p, xi_n, xf_n, emb, W1, b1, Vf, Vh, yfm);
    k_padh1<<<B2/256, 256, 0, stream>>>(H1h);

    const int mblocks = B2 / (16 * MB);          // 512
    const int ntiles  = HIDDEN / 16;             // 25
    const int blocks  = (mblocks * ntiles) / 8;  // 1600, exact -> full waves only
    k_gemm_swish<<<blocks, 256, 0, stream>>>(Vh, KV, W2h, KV, b2eff, H1h, KH, KV/32, ntiles, mblocks);
    k_gemm_swish<<<blocks, 256, 0, stream>>>(H1h, KH, W3h, KH, b3,    H2h, HIDDEN, KH/32, ntiles, mblocks);

    k_final<<<BATCH/256, 256, 0, stream>>>(H2h, yfm, W4, b4, out);
}
